// MultiHeadAttention_77163382440533
// MI455X (gfx1250) — compile-verified, matrix operand order checked by reading, not run on hardware
//
#include <hip/hip_runtime.h>
#include <hip/hip_bf16.h>

// ---------------------------------------------------------------------------
// MultiHeadAttention forward, CDNA5 (gfx1250), bf16 WMMA pipeline.
//   B=2, S=2048, D=1024, H=16, d_k=64.
// ~69 GFLOP vs ~90 MB minimum traffic -> strongly compute bound:
// route all matmuls through v_wmma_f32_16x16x32_bf16 (fp32 accumulate).
// Weights converted to bf16 once; GEMM waves own 16x128 tiles (8 WMMA per
// K-step per A-fragment load). Attention is flash-style (no S x S tensor).
// ---------------------------------------------------------------------------

#define D_MODEL 1024
#define N_HEADS 16
#define D_K     64
#define BATCH   2
#define SEQ     2048
#define ROWS    (BATCH * SEQ)          // 4096 token rows

typedef __bf16 bf16_t;
typedef __attribute__((ext_vector_type(16))) __bf16 v16bf;
typedef __attribute__((ext_vector_type(8)))  __bf16 v8bf;
typedef __attribute__((ext_vector_type(4)))  __bf16 v4bf;
typedef __attribute__((ext_vector_type(8)))  float  v8f;
typedef __attribute__((ext_vector_type(4)))  float  v4f;

// --- WMMA A/B fragment loaders (16x32 bf16 tile, wave32 layout) -------------
// ISA layout: lanes 0-15  -> row = lane,    K = {0..7, 16..23}
//             lanes 16-31 -> row = lane-16, K = {8..15, 24..31}
// `p` must already include row*ld + kofs (kofs = 0 or 8 by lane half), so the
// two halves are contiguous 16-byte runs -> b128 loads.
__device__ __forceinline__ v16bf frag_ld_bf16(const bf16_t* p) {
  v8bf lo = *(const v8bf*)p;          // K kofs+0..7
  v8bf hi = *(const v8bf*)(p + 16);   // K kofs+16..23
  return __builtin_shufflevector(lo, hi, 0, 1, 2, 3, 4, 5, 6, 7,
                                         8, 9, 10, 11, 12, 13, 14, 15);
}

__device__ __forceinline__ v16bf frag_ld_f32(const float* p) {
  v4f a0 = *(const v4f*)(p + 0);
  v4f a1 = *(const v4f*)(p + 4);
  v4f b0 = *(const v4f*)(p + 16);
  v4f b1 = *(const v4f*)(p + 20);
  v16bf a;
#pragma unroll
  for (int i = 0; i < 4; ++i) {
    a[i]      = (__bf16)a0[i];
    a[4 + i]  = (__bf16)a1[i];
    a[8 + i]  = (__bf16)b0[i];
    a[12 + i] = (__bf16)b1[i];
  }
  return a;
}

__device__ __forceinline__ v8f wmma_bf16(v16bf a, v16bf b, v8f c) {
  return __builtin_amdgcn_wmma_f32_16x16x32_bf16(
      /*neg_a=*/false, a, /*neg_b=*/false, b,
      /*c_mod=*/(short)0, c, /*reuse_a=*/false, /*reuse_b=*/false);
}

// ---------------------------------------------------------------------------
// Kernel 0: one-shot fp32 -> bf16 weight conversion (Wq,Wk,Wv,Wo).
// Keeps all cvt out of the GEMM hot loops.
// ---------------------------------------------------------------------------
__global__ void mha_cvt_weights_kernel(const float* __restrict__ Wq,
                                       const float* __restrict__ Wk,
                                       const float* __restrict__ Wv,
                                       const float* __restrict__ Wo,
                                       bf16_t* __restrict__ w_bf) {
  const int mat = blockIdx.y;
  const float* W = (mat == 0) ? Wq : (mat == 1) ? Wk : (mat == 2) ? Wv : Wo;
  bf16_t* dst = w_bf + (size_t)mat * D_MODEL * D_MODEL;
  const int i = (blockIdx.x * 256 + threadIdx.x) * 4;
  v4f x = *(const v4f*)(W + i);
  v4bf y;
#pragma unroll
  for (int j = 0; j < 4; ++j) y[j] = (__bf16)x[j];
  *(v4bf*)(dst + i) = y;
}

// ---------------------------------------------------------------------------
// Kernel 1: fused QKV projections. y = x @ W^T + b, x fp32 [4096,1024],
// W bf16 [1024,1024]. One wave owns a 16x128 output tile: per 32-K step,
// 1 A-fragment (fp32->bf16, reused 8x) + 8 bf16 B-fragments -> 8 WMMAs.
// q,k written bf16 as [B,H,S,64]; v written bf16 transposed [B,H,64,S].
// ---------------------------------------------------------------------------
__global__ void mha_qkv_proj_kernel(const float* __restrict__ Q,
                                    const float* __restrict__ K,
                                    const float* __restrict__ V,
                                    const bf16_t* __restrict__ w_bf,
                                    const float* __restrict__ bq,
                                    const float* __restrict__ bk,
                                    const float* __restrict__ bv,
                                    bf16_t* __restrict__ q_bf,
                                    bf16_t* __restrict__ k_bf,
                                    bf16_t* __restrict__ v_tb) {
  const int lane = threadIdx.x & 31;
  const int wave = threadIdx.x >> 5;
  const int tile = blockIdx.x * 8 + wave;   // 2048 tiles: 256 M x 8 N-quads
  const int tm   = tile >> 3;               // 16-row tile   [0,256)
  const int tq   = tile & 7;                // 128-col tile  [0,8)
  const int mat  = blockIdx.y;              // 0=q 1=k 2=v

  const float* X = (mat == 0) ? Q : (mat == 1) ? K : V;
  const float* bias = (mat == 0) ? bq : (mat == 1) ? bk : bv;
  const bf16_t* W = w_bf + (size_t)mat * D_MODEL * D_MODEL;

  const int r    = lane & 15;
  const int kofs = (lane < 16) ? 0 : 8;
  const float*  Ap = X + (tm * 16 + r) * D_MODEL + kofs;
  const bf16_t* Bp = W + (tq * 128 + r) * D_MODEL + kofs;

  v8f acc[8] = {};
  for (int k0 = 0; k0 < D_MODEL; k0 += 32) {
    const v16bf a = frag_ld_f32(Ap + k0);
#pragma unroll
    for (int j = 0; j < 8; ++j) {
      const v16bf b = frag_ld_bf16(Bp + j * 16 * D_MODEL + k0);
      acc[j] = wmma_bf16(a, b, acc[j]);
    }
  }

  const int mlo = tm * 16 + ((lane < 16) ? 0 : 8);
#pragma unroll
  for (int j = 0; j < 8; ++j) {
    const int n = tq * 128 + j * 16 + r;
    const float bn = bias[n];
    const int h = n >> 6;                    // head
    const int d = n & 63;                    // dim inside head
#pragma unroll
    for (int rr = 0; rr < 8; ++rr) {
      const float val = acc[j][rr] + bn;
      const int m = mlo + rr;                // token row in [0,4096)
      const int bb = m >> 11;                // batch
      const int ss = m & 2047;               // seq position
      const int bh = bb * N_HEADS + h;
      if (mat == 2) {
        v_tb[(bh * D_K + d) * SEQ + ss] = (__bf16)val;   // [B,H,64,S]
      } else if (mat == 1) {
        k_bf[(bh * SEQ + ss) * D_K + d] = (__bf16)val;   // [B,H,S,64]
      } else {
        q_bf[(bh * SEQ + ss) * D_K + d] = (__bf16)val;
      }
    }
  }
}

// ---------------------------------------------------------------------------
// Kernel 2: flash attention. One wave = one (b,h, 16-query) tile, streaming
// 32-key blocks with online softmax. P tile staged through LDS to convert
// C-layout -> A-layout. Output bf16 [B*S, 1024] (token-major).
// ---------------------------------------------------------------------------
__global__ void mha_flash_attn_kernel(const bf16_t* __restrict__ q_bf,
                                      const bf16_t* __restrict__ k_bf,
                                      const bf16_t* __restrict__ v_tb,
                                      bf16_t* __restrict__ attn_bf) {
  __shared__ __bf16 lds_p[8][16 * 32];       // per-wave P staging, 8 KB

  const int lane = threadIdx.x & 31;
  const int wave = threadIdx.x >> 5;
  const int gw   = blockIdx.x * 8 + wave;    // 4096 waves total
  const int bh   = gw >> 7;                  // (b*16+h) in [0,32)
  const int q0   = (gw & 127) * 16;          // query tile origin

  const int r    = lane & 15;
  const int kofs = (lane < 16) ? 0 : 8;
  const int half = (lane < 16) ? 0 : 8;

  const bf16_t* qbase = q_bf + (bh * SEQ + q0) * D_K;
  const bf16_t* kmat  = k_bf + bh * SEQ * D_K;
  const bf16_t* vmat  = v_tb + bh * D_K * SEQ;
  bf16_t* lds = (bf16_t*)&lds_p[wave][0];

  // Cache the 16x64 query tile as two A fragments (K = 0..31, 32..63).
  const v16bf qa0 = frag_ld_bf16(qbase + r * D_K + kofs);
  const v16bf qa1 = frag_ld_bf16(qbase + r * D_K + kofs + 32);

  v8f acc0 = {}, acc1 = {}, acc2 = {}, acc3 = {};
  float m_run[8], l_run[8];
#pragma unroll
  for (int i = 0; i < 8; ++i) { m_run[i] = -1e30f; l_run[i] = 0.0f; }

  const float scale = 0.125f;                // 1/sqrt(64)

  for (int kb = 0; kb < SEQ; kb += 32) {
    // prefetch next key/value blocks into cache (global_prefetch_b8)
    if (kb + 32 < SEQ) {
      __builtin_prefetch(kmat + (kb + 32 + r) * D_K, 0, 0);
      __builtin_prefetch(vmat + r * SEQ + kb + 32, 0, 0);
    }

    // ---- scores: two 16x16 tiles, K = d_k = 64 (2 WMMAs each) ----
    v8f s0 = {}, s1 = {};
    s0 = wmma_bf16(qa0, frag_ld_bf16(kmat + (kb + r) * D_K + kofs),       s0);
    s0 = wmma_bf16(qa1, frag_ld_bf16(kmat + (kb + r) * D_K + kofs + 32),  s0);
    s1 = wmma_bf16(qa0, frag_ld_bf16(kmat + (kb + 16 + r) * D_K + kofs),      s1);
    s1 = wmma_bf16(qa1, frag_ld_bf16(kmat + (kb + 16 + r) * D_K + kofs + 32), s1);
    s0 = s0 * scale;
    s1 = s1 * scale;

    // ---- online softmax per row (rows live across 16-lane halves) ----
#pragma unroll
    for (int rr = 0; rr < 8; ++rr) {
      float rmax = fmaxf(s0[rr], s1[rr]);
#pragma unroll
      for (int off = 8; off > 0; off >>= 1)
        rmax = fmaxf(rmax, __shfl_xor(rmax, off, 16));
      const float m_new   = fmaxf(m_run[rr], rmax);
      const float rescale = __expf(m_run[rr] - m_new);
      const float p0 = __expf(s0[rr] - m_new);
      const float p1 = __expf(s1[rr] - m_new);
      float rsum = p0 + p1;
#pragma unroll
      for (int off = 8; off > 0; off >>= 1)
        rsum += __shfl_xor(rsum, off, 16);
      l_run[rr] = l_run[rr] * rescale + rsum;
      m_run[rr] = m_new;
      acc0[rr] *= rescale; acc1[rr] *= rescale;
      acc2[rr] *= rescale; acc3[rr] *= rescale;
      // stage P (C layout) into LDS as a row-major 16x32 bf16 tile
      const int row = rr + half;
      lds[row * 32 + r]      = (__bf16)p0;
      lds[row * 32 + 16 + r] = (__bf16)p1;
    }
    asm volatile("s_wait_dscnt 0" ::: "memory");   // LDS RAW within wave

    // ---- P (16x32) as A fragment; V block (32x64) as four B fragments ----
    const v16bf pa = frag_ld_bf16(lds + r * 32 + kofs);
    acc0 = wmma_bf16(pa, frag_ld_bf16(vmat + ( 0 + r) * SEQ + kb + kofs), acc0);
    acc1 = wmma_bf16(pa, frag_ld_bf16(vmat + (16 + r) * SEQ + kb + kofs), acc1);
    acc2 = wmma_bf16(pa, frag_ld_bf16(vmat + (32 + r) * SEQ + kb + kofs), acc2);
    acc3 = wmma_bf16(pa, frag_ld_bf16(vmat + (48 + r) * SEQ + kb + kofs), acc3);
    asm volatile("s_wait_dscnt 0" ::: "memory");   // done reading LDS tile
  }

  // ---- normalize and write bf16 output [B*S, H*64] ----
  const int b = bh >> 4, h = bh & 15;
#pragma unroll
  for (int rr = 0; rr < 8; ++rr) {
    const float inv = 1.0f / l_run[rr];
    const int row = b * SEQ + q0 + rr + half;
    const int col = h * D_K + r;
    attn_bf[row * D_MODEL + col]      = (__bf16)(acc0[rr] * inv);
    attn_bf[row * D_MODEL + col + 16] = (__bf16)(acc1[rr] * inv);
    attn_bf[row * D_MODEL + col + 32] = (__bf16)(acc2[rr] * inv);
    attn_bf[row * D_MODEL + col + 48] = (__bf16)(acc3[rr] * inv);
  }
}

// ---------------------------------------------------------------------------
// Kernel 3: output projection. out = attn(bf16) @ Wo_bf^T + bo, fp32 result.
// Same 16x128-per-wave tiling as kernel 1, all-bf16 operand loads.
// ---------------------------------------------------------------------------
__global__ void mha_out_proj_kernel(const bf16_t* __restrict__ attn_bf,
                                    const bf16_t* __restrict__ Wo_bf,
                                    const float* __restrict__ bo,
                                    float* __restrict__ out) {
  const int lane = threadIdx.x & 31;
  const int wave = threadIdx.x >> 5;
  const int tile = blockIdx.x * 8 + wave;   // 2048 tiles
  const int tm   = tile >> 3;
  const int tq   = tile & 7;

  const int r    = lane & 15;
  const int kofs = (lane < 16) ? 0 : 8;
  const bf16_t* Ap = attn_bf + (tm * 16 + r) * D_MODEL + kofs;
  const bf16_t* Bp = Wo_bf + (tq * 128 + r) * D_MODEL + kofs;

  v8f acc[8] = {};
  for (int k0 = 0; k0 < D_MODEL; k0 += 32) {
    const v16bf a = frag_ld_bf16(Ap + k0);
#pragma unroll
    for (int j = 0; j < 8; ++j) {
      const v16bf b = frag_ld_bf16(Bp + j * 16 * D_MODEL + k0);
      acc[j] = wmma_bf16(a, b, acc[j]);
    }
  }

  const int mlo = tm * 16 + ((lane < 16) ? 0 : 8);
#pragma unroll
  for (int j = 0; j < 8; ++j) {
    const int n = tq * 128 + j * 16 + r;
    const float bn = bo[n];
#pragma unroll
    for (int rr = 0; rr < 8; ++rr)
      out[(mlo + rr) * D_MODEL + n] = acc[j][rr] + bn;
  }
}

// ---------------------------------------------------------------------------
extern "C" void kernel_launch(void* const* d_in, const int* in_sizes, int n_in,
                              void* d_out, int out_size, void* d_ws, size_t ws_size,
                              hipStream_t stream) {
  const float* Q  = (const float*)d_in[0];
  const float* K  = (const float*)d_in[1];
  const float* V  = (const float*)d_in[2];
  const float* Wq = (const float*)d_in[3];
  const float* bq = (const float*)d_in[4];
  const float* Wk = (const float*)d_in[5];
  const float* bk = (const float*)d_in[6];
  const float* Wv = (const float*)d_in[7];
  const float* bv = (const float*)d_in[8];
  const float* Wo = (const float*)d_in[9];
  const float* bo = (const float*)d_in[10];
  float* out = (float*)d_out;

  // workspace layout (bf16 elems): w[4M] q[4M] k[4M] v_t[4M] attn[4M] = 40 MB
  const size_t wElems  = (size_t)D_MODEL * D_MODEL;   // 1M per weight matrix
  const size_t mElems  = (size_t)ROWS * D_MODEL;      // 4M per activation
  bf16_t* w_bf   = (bf16_t*)d_ws;                     // Wq|Wk|Wv|Wo bf16
  bf16_t* q_bf   = w_bf + 4 * wElems;
  bf16_t* k_bf   = q_bf + mElems;
  bf16_t* v_tb   = k_bf + mElems;
  bf16_t* attnbf = v_tb + mElems;

  // 0) convert 4 weight matrices to bf16 (4 x 1M elems, 4 floats/thread)
  mha_cvt_weights_kernel<<<dim3(1024, 4), 256, 0, stream>>>(Wq, Wk, Wv, Wo, w_bf);

  // 1) QKV projections: 2048 wave-tiles per matrix, 8 waves/block
  mha_qkv_proj_kernel<<<dim3(256, 3), 256, 0, stream>>>(
      Q, K, V, w_bf, bq, bk, bv, q_bf, k_bf, v_tb);

  // 2) attention: 2*16*(2048/16) = 4096 waves -> 512 blocks of 8 waves
  mha_flash_attn_kernel<<<dim3(512), 256, 0, stream>>>(q_bf, k_bf, v_tb, attnbf);

  // 3) output projection: 2048 wave-tiles
  mha_out_proj_kernel<<<dim3(256), 256, 0, stream>>>(attnbf, w_bf + 3 * wElems, bo, out);
}
